// LSTM_44092134260750
// MI455X (gfx1250) — matrix operand order repaired and checked by reading, not compiled, not run on hardware
//
#include <hip/hip_runtime.h>

typedef __attribute__((ext_vector_type(16))) __bf16 v16bf;
typedef __attribute__((ext_vector_type(8)))  __bf16 v8bf;
typedef __attribute__((ext_vector_type(4)))  __bf16 v4bf;
typedef __attribute__((ext_vector_type(8)))  float  v8f;

#define DDIM    1024
#define HDIM    1024
#define NBATCH  64
#define TSTEPS  512
#define NBLOCKS 64      // one block per hidden 16-col tile j
#define NTHREADS 256    // 8 waves = 4 batch-tiles (m) x 2 K-halves

// ---------------------------------------------------------------------------
// Pack W (K=1024 x N=4096, row-major fp32) into WMMA B-fragment order, bf16:
// P[kc][nt][lane][i]: lane holds column nt*16+(l&15), K vals kc*32+(l>>4)*16+i.
// ---------------------------------------------------------------------------
__global__ void pack_w_kernel(const float* __restrict__ W, __bf16* __restrict__ P) {
  int idx  = blockIdx.x * blockDim.x + threadIdx.x;   // 0 .. 4M-1
  int i    = idx & 15;
  int lane = (idx >> 4) & 31;
  int nt   = (idx >> 9) & 255;
  int kc   = idx >> 17;
  int k = kc * 32 + ((lane >> 4) << 4) + i;
  int n = (nt << 4) + (lane & 15);
  P[idx] = (__bf16)W[(size_t)k * 4096 + n];
}

// x fp32 -> bf16, row-major (one-time; removes all per-step A-side converts)
__global__ void pack_x_kernel(const float* __restrict__ X, __bf16* __restrict__ P) {
  int idx = blockIdx.x * blockDim.x + threadIdx.x;    // 0 .. 8M-1 (x4 floats)
  float4 f = ((const float4*)X)[idx];
  v4bf o;
  o[0] = (__bf16)f.x; o[1] = (__bf16)f.y; o[2] = (__bf16)f.z; o[3] = (__bf16)f.w;
  *(v4bf*)(P + (size_t)idx * 4) = o;
}

__global__ void init_kernel(const float* __restrict__ h0, __bf16* __restrict__ hbuf,
                            unsigned* cnt, unsigned* gen) {
  int idx = blockIdx.x * blockDim.x + threadIdx.x;
  if (idx < NBATCH * HDIM) hbuf[idx] = (__bf16)h0[idx];
  if (idx == 0) { *cnt = 0u; *gen = 0u; }
}

__device__ __forceinline__ float sigmoidf_fast(float v) {
  return 1.0f / (1.0f + __expf(-v));
}

__device__ __forceinline__ void grid_barrier(unsigned* cnt, unsigned* gen) {
  __syncthreads();
  if (threadIdx.x == 0) {
    __threadfence();  // make h/out stores device-visible
    unsigned g = __hip_atomic_load(gen, __ATOMIC_RELAXED, __HIP_MEMORY_SCOPE_AGENT);
    unsigned v = __hip_atomic_fetch_add(cnt, 1u, __ATOMIC_ACQ_REL, __HIP_MEMORY_SCOPE_AGENT);
    if (v == NBLOCKS - 1) {
      __hip_atomic_store(cnt, 0u, __ATOMIC_RELAXED, __HIP_MEMORY_SCOPE_AGENT);
      __hip_atomic_fetch_add(gen, 1u, __ATOMIC_ACQ_REL, __HIP_MEMORY_SCOPE_AGENT);
    } else {
      while (__hip_atomic_load(gen, __ATOMIC_ACQUIRE, __HIP_MEMORY_SCOPE_AGENT) == g) {
        __builtin_amdgcn_s_sleep(2);
      }
    }
  }
  __syncthreads();
}

// ---------------------------------------------------------------------------
// Persistent LSTM. Block j caches its full B working set (4 gates x 64
// K-chunks = 256 KB bf16) in LDS ONCE; all 512 steps read B via ds_load.
// wave w: m = w&3 (batch tile), khalf = w>>2 (0: x@Wx + bias, 1: h@Wh).
// Partials reduced via 16 KB LDS; c state lives in VGPRs for all 512 steps.
// ---------------------------------------------------------------------------
__global__ __launch_bounds__(NTHREADS, 1)
void lstm_persistent_kernel(const __bf16* __restrict__ xbf,
                            const float* __restrict__ bias,
                            const __bf16* __restrict__ WxP,
                            const __bf16* __restrict__ WhP,
                            __bf16* __restrict__ hb0,
                            __bf16* __restrict__ hb1,
                            float* __restrict__ out,
                            unsigned* cnt, unsigned* gen) {
  extern __shared__ char smem[];
  __bf16* Blds = (__bf16*)smem;                 // 256 KB: [kc64][g4][lane32][16]
  float*  Pred = (float*)(smem + 256 * 1024);   // 16 KB:  [m4][g4][lane32][8]

  const int lane  = threadIdx.x & 31;
  const int wave  = threadIdx.x >> 5;
  const int m     = wave & 3;       // batch tile 0..3
  const int khalf = wave >> 2;      // 0 = x half, 1 = h half
  const int j     = blockIdx.x;     // hidden tile 0..63
  const int l16   = lane & 15;
  const int hi    = lane >> 4;
  const int col   = j * 16 + l16;
  const int arow  = m * 16 + l16;
  const int aoff  = hi ? 16 : 0;

  // ---- One-time LDS fill of the B working set (16B units, coalesced) ----
  for (int u = threadIdx.x; u < 64 * 4 * 32 * 2; u += NTHREADS) {
    int half = u & 1;
    int ln   = (u >> 1) & 31;
    int g    = (u >> 6) & 3;
    int kc   = u >> 8;
    const __bf16* src = (kc < 32)
        ? (WxP + (((size_t)kc        * 256 + (g * 64 + j)) * 32 + ln) * 16)
        : (WhP + (((size_t)(kc - 32) * 256 + (g * 64 + j)) * 32 + ln) * 16);
    *(uint4*)(Blds + (size_t)u * 8) = *(const uint4*)(src + half * 8);
  }

  float bg[4];
#pragma unroll
  for (int g = 0; g < 4; ++g) bg[g] = bias[g * HDIM + col];

  v8f c = {0.f, 0.f, 0.f, 0.f, 0.f, 0.f, 0.f, 0.f};

  __syncthreads();

#pragma unroll 1
  for (int t = 0; t < TSTEPS; ++t) {
    const __bf16* hcur  = (t & 1) ? hb1 : hb0;
    __bf16*       hnext = (t & 1) ? hb0 : hb1;

    v8f acc[4];
#pragma unroll
    for (int g = 0; g < 4; ++g) {
      float b0 = khalf ? 0.f : bg[g];
      v8f z = {b0, b0, b0, b0, b0, b0, b0, b0};
      acc[g] = z;
    }

    const __bf16* Arow = khalf ? (hcur + (size_t)arow * HDIM)
                               : (xbf + ((size_t)arow * TSTEPS + t) * DDIM);

#pragma unroll 4
    for (int kc = 0; kc < 32; ++kc) {
      const int kbase = kc * 32 + aoff;
      // A fragment (contiguous layout): K = {kbase..+15}, pure bf16 loads
      v8bf a0 = *(const v8bf*)(Arow + kbase);
      v8bf a1 = *(const v8bf*)(Arow + kbase + 8);
      v16bf a;
#pragma unroll
      for (int e = 0; e < 8; ++e) { a[e] = a0[e]; a[8 + e] = a1[e]; }

      const __bf16* bb = Blds + ((size_t)(khalf * 32 + kc) * 4 * 32 + lane) * 16;
#pragma unroll
      for (int g = 0; g < 4; ++g) {
        v16bf bf = *(const v16bf*)(bb + (size_t)g * (32 * 16));   // ds_load
        acc[g] = __builtin_amdgcn_wmma_f32_16x16x32_bf16(
            false, a, false, bf, (short)0, acc[g], false, false);
      }
    }

    // ---- cross-wave K reduction through LDS ----
    if (khalf == 1) {
#pragma unroll
      for (int g = 0; g < 4; ++g)
        *(v8f*)(Pred + (((size_t)m * 4 + g) * 32 + lane) * 8) = acc[g];
    }
    __syncthreads();

    if (khalf == 0) {
#pragma unroll
      for (int g = 0; g < 4; ++g) {
        v8f p = *(const v8f*)(Pred + (((size_t)m * 4 + g) * 32 + lane) * 8);
        acc[g] += p;
      }
      // Gate math + state update (C/D layout: row = m*16 + v + 8*hi)
#pragma unroll
      for (int v = 0; v < 8; ++v) {
        float iv = sigmoidf_fast(acc[0][v]);
        float fv = sigmoidf_fast(acc[1][v]);
        float ov = sigmoidf_fast(acc[2][v]);
        float gv = tanhf(acc[3][v]);
        float cn = fv * c[v] + iv * gv;
        c[v] = cn;
        float hn = ov * tanhf(cn);
        const int row = m * 16 + v + hi * 8;
        hnext[(size_t)row * HDIM + col]              = (__bf16)hn;
        out[((size_t)row * TSTEPS + t) * HDIM + col] = hn;
      }
    }

    grid_barrier(cnt, gen);
  }
}

extern "C" void kernel_launch(void* const* d_in, const int* in_sizes, int n_in,
                              void* d_out, int out_size, void* d_ws, size_t ws_size,
                              hipStream_t stream) {
  (void)in_sizes; (void)n_in; (void)out_size; (void)ws_size;
  const float* x  = (const float*)d_in[0];   // (64, 512, 1024)
  const float* h0 = (const float*)d_in[1];   // (64, 1024)
  const float* Wx = (const float*)d_in[2];   // (1024, 4096)
  const float* Wh = (const float*)d_in[3];   // (1024, 4096)
  const float* b  = (const float*)d_in[4];   // (4096,)
  float* out = (float*)d_out;                // (64, 512, 1024)

  char* ws = (char*)d_ws;
  __bf16* xbf = (__bf16*)ws;                                    // 64 MB
  __bf16* WxP = (__bf16*)(ws + (size_t)64 * 1024 * 1024);       // 8 MB
  __bf16* WhP = (__bf16*)(ws + (size_t)72 * 1024 * 1024);       // 8 MB
  __bf16* hb0 = (__bf16*)(ws + (size_t)80 * 1024 * 1024);       // 128 KB
  __bf16* hb1 = hb0 + NBATCH * HDIM;                            // 128 KB
  unsigned* cnt = (unsigned*)(hb1 + NBATCH * HDIM);
  unsigned* gen = cnt + 64;  // separate cachelines

  const int packW = 1024 * 4096;
  pack_w_kernel<<<packW / 256, 256, 0, stream>>>(Wx, WxP);
  pack_w_kernel<<<packW / 256, 256, 0, stream>>>(Wh, WhP);
  const int packX4 = NBATCH * TSTEPS * DDIM / 4;
  pack_x_kernel<<<packX4 / 256, 256, 0, stream>>>(x, xbf);
  init_kernel<<<(NBATCH * HDIM + 255) / 256, 256, 0, stream>>>(h0, hb0, cnt, gen);

  const size_t ldsBytes = 256 * 1024 + 16 * 1024;   // B cache + partial buffer
  lstm_persistent_kernel<<<NBLOCKS, NTHREADS, ldsBytes, stream>>>(
      xbf, b, WxP, WhP, hb0, hb1, out, cnt, gen);
}
